// LSTM_residual_83932250898884
// MI455X (gfx1250) — compile-verified, matrix operand order checked by reading, not run on hardware
//
#include <hip/hip_runtime.h>
#include <stdint.h>

typedef unsigned short u16;
typedef __attribute__((ext_vector_type(16))) __bf16 bf16x16;
typedef __attribute__((ext_vector_type(8)))  float  f32x8;
typedef int v4i_ __attribute__((vector_size(16)));

#define BB   32
#define TT   512
#define G4   2048
#define MMR  (BB*TT)          /* 16384 rows */
#define NBLK 5
#define LEPS 1e-5f

#if __has_builtin(__builtin_amdgcn_global_load_async_to_lds_b128)
#define HAVE_ASYNC_LDS 1
#else
#define HAVE_ASYNC_LDS 0
#endif

__device__ __forceinline__ u16 f2bf(float f){
  unsigned u = __float_as_uint(f);
  unsigned r = (u + 0x7FFFu + ((u >> 16) & 1u)) >> 16;   // RNE
  return (u16)r;
}
__device__ __forceinline__ float bf2f(u16 h){ return __uint_as_float(((unsigned)h) << 16); }

union FragU { uint4 u[2]; bf16x16 v; };
__device__ __forceinline__ bf16x16 mkfrag(uint4 a, uint4 b){ FragU x; x.u[0]=a; x.u[1]=b; return x.v; }
__device__ __forceinline__ f32x8 zero8(){ f32x8 z; for (int i=0;i<8;++i) z[i]=0.f; return z; }
__device__ __forceinline__ float sigmoidf_(float x){ return 1.f/(1.f+__expf(-x)); }

// 16B global -> LDS copy: async DMA on gfx1250 if the builtin exists.
// Builtin signature (from hipcc diag): param0 = int4 * in device (AS1) space.
__device__ __forceinline__ void cp16B(const u16* g, u16* l){
#if HAVE_ASYNC_LDS
  __builtin_amdgcn_global_load_async_to_lds_b128(
      (__attribute__((address_space(1))) v4i_*)(uintptr_t)g,
      (__attribute__((address_space(3))) v4i_*)(unsigned)(uintptr_t)l,
      0, 0);
#else
  *(uint4*)l = *(const uint4*)g;
#endif
}
template<int N>
__device__ __forceinline__ void wait_async(){
#if HAVE_ASYNC_LDS
 #if __has_builtin(__builtin_amdgcn_s_wait_asynccnt)
  __builtin_amdgcn_s_wait_asynccnt((unsigned short)N);
 #else
  asm volatile("s_wait_asynccnt %0" :: "i"(N) : "memory");
 #endif
#endif
}

// ---------------------------------------------------------------- conversions
__global__ void cvt_bf16(const float* __restrict__ src, u16* __restrict__ dst, size_t n){
  for (size_t i = blockIdx.x*(size_t)blockDim.x + threadIdx.x; i < n;
       i += (size_t)gridDim.x*blockDim.x)
    dst[i] = f2bf(src[i]);
}

// B-matrix [K][N] (row-major f32) -> fragment-swizzled bf16.
// frag element idx = ((nt*KT + kt)*32 + lane)*16 + e
// lane: n_in = lane&15, half = lane>>4 ; e<8 -> k_in = half*8+e, else 16+half*8+(e-8)
__global__ void pack_bfrag(const float* __restrict__ src, u16* __restrict__ dst,
                           int K, int Nsrc, int Npad){
  const int KT = K >> 5;
  size_t total = (size_t)K * Npad;
  for (size_t idx = blockIdx.x*(size_t)blockDim.x + threadIdx.x; idx < total;
       idx += (size_t)gridDim.x*blockDim.x){
    int e    = (int)(idx & 15);
    int lane = (int)((idx >> 4) & 31);
    size_t tp = idx >> 9;
    int kt = (int)(tp % KT), nt = (int)(tp / KT);
    int half = lane >> 4, n_in = lane & 15;
    int k_in = (e < 8) ? (half*8 + e) : (16 + half*8 + (e - 8));
    int kk  = kt*32 + k_in;
    int col = nt*16 + n_in;
    float v = (col < Nsrc) ? src[(size_t)kk*Nsrc + col] : 0.f;
    dst[idx] = f2bf(v);
  }
}

// Wh [512][2048] -> LSTM-custom fragment swizzle: tile tj = wg*8+wave packs gate
// columns col = 512*g + tj*4 + q  with n_in = g*4+q  (i/f/g/o of 4 h-cols per tile)
__global__ void pack_whfrag(const float* __restrict__ Wh, u16* __restrict__ dst){
  const size_t total = (size_t)128*16*32*16;   // 1,048,576
  for (size_t idx = blockIdx.x*(size_t)blockDim.x + threadIdx.x; idx < total;
       idx += (size_t)gridDim.x*blockDim.x){
    int e    = (int)(idx & 15);
    int lane = (int)((idx >> 4) & 31);
    size_t tp = idx >> 9;
    int kt = (int)(tp & 15), tj = (int)(tp >> 4);
    int half = lane >> 4, n_in = lane & 15, g = n_in >> 2, q = n_in & 3;
    int k_in = (e < 8) ? (half*8 + e) : (16 + half*8 + (e - 8));
    int kk  = kt*32 + k_in;
    int col = g*512 + tj*4 + q;
    dst[idx] = f2bf(Wh[(size_t)kk*G4 + col]);
  }
}

// ---------------------------------------------------------------- concat packs
__global__ void pack_cat3(const u16* __restrict__ xb, const u16* __restrict__ y,
                          const u16* __restrict__ yl, u16* __restrict__ cat){
  size_t total = (size_t)MMR * 1088;
  for (size_t i = blockIdx.x*(size_t)blockDim.x + threadIdx.x; i < total;
       i += (size_t)gridDim.x*blockDim.x){
    size_t m = i / 1088; int c = (int)(i % 1088);
    u16 v;
    if      (c < 64)  v = xb[m*64 + c];
    else if (c < 576) v = y [m*512 + (c - 64)];
    else              v = yl[m*512 + (c - 576)];
    cat[i] = v;
  }
}
__global__ void pack_cat2(const u16* __restrict__ y, const u16* __restrict__ yl,
                          u16* __restrict__ cat){
  size_t total = (size_t)MMR * 1024;
  for (size_t i = blockIdx.x*(size_t)blockDim.x + threadIdx.x; i < total;
       i += (size_t)gridDim.x*blockDim.x){
    size_t m = i >> 10; int c = (int)(i & 1023);
    cat[i] = (c < 512) ? y[m*512 + c] : yl[m*512 + (c - 512)];
  }
}

// ---------------------------------------------------------------- big GEMM
// out[M,N](bf16) = act(A[M,K](bf16) @ Bfrag + bias) ; optional pre-act store.
// WG = 256 thr (8 waves), tile 128x64, wave tile 32x32 (2x2 WMMA accs), KC=32.
// A tiles double-buffered in LDS, staged by async global->LDS DMA when available.
__global__ __launch_bounds__(256) void gemm_wmma(
    const u16* __restrict__ A, const u16* __restrict__ Bf,
    const float* __restrict__ bias,
    u16* __restrict__ out, u16* __restrict__ out_pre,
    int N, int K, int relu)
{
  __shared__ u16 As[2][128*40];
  const int KT = K >> 5;
  const int bn = blockIdx.x * 64, bm = blockIdx.y * 128;
  const int tid = threadIdx.x, wave = tid >> 5, lane = tid & 31;
  const int wm = wave & 3, wn = wave >> 2;
  const int n = lane & 15, half = lane >> 4;

  f32x8 acc[2][2];
  for (int mt = 0; mt < 2; ++mt) for (int nt = 0; nt < 2; ++nt) acc[mt][nt] = zero8();

  const int row0 = tid >> 2,          seg0 = tid & 3;          // chunk tid
  const int row1 = (tid + 256) >> 2,  seg1 = tid & 3;          // chunk tid+256

  // prologue: stage kt=0 into buffer 0
  cp16B(&A[(size_t)(bm + row0)*K + seg0*8], &As[0][row0*40 + seg0*8]);
  cp16B(&A[(size_t)(bm + row1)*K + seg1*8], &As[0][row1*40 + seg1*8]);

  int cur = 0;
  for (int kt = 0; kt < KT; ++kt){
    if (kt + 1 < KT){
      int nxt = cur ^ 1;
      size_t kb = (size_t)(kt + 1)*32;
      cp16B(&A[(size_t)(bm + row0)*K + kb + seg0*8], &As[nxt][row0*40 + seg0*8]);
      cp16B(&A[(size_t)(bm + row1)*K + kb + seg1*8], &As[nxt][row1*40 + seg1*8]);
      wait_async<2>();            // retire the older (current) stage only
      __builtin_prefetch(A + (size_t)(bm + (tid >> 1))*K + kb + 32, 0, 1);
    } else {
      wait_async<0>();
    }
    __syncthreads();              // As[cur] visible to every wave

    bf16x16 af[2], bfr[2];
    #pragma unroll
    for (int mt = 0; mt < 2; ++mt){
      int lrow = wm*32 + mt*16 + n;
      int base = lrow*40 + half*8;
      af[mt] = mkfrag(*(const uint4*)&As[cur][base], *(const uint4*)&As[cur][base + 16]);
    }
    #pragma unroll
    for (int nt = 0; nt < 2; ++nt){
      size_t bo = (((size_t)((bn >> 4) + wn*2 + nt))*KT + kt)*512 + (size_t)lane*16;
      const uint4* bp = (const uint4*)(Bf + bo);
      bfr[nt] = mkfrag(bp[0], bp[1]);
    }
    #pragma unroll
    for (int mt = 0; mt < 2; ++mt)
      #pragma unroll
      for (int nt = 0; nt < 2; ++nt)
        acc[mt][nt] = __builtin_amdgcn_wmma_f32_16x16x32_bf16(
            false, af[mt], false, bfr[nt], (short)0, acc[mt][nt], false, false);
    __syncthreads();              // everyone done reading As[cur] before overwrite
    cur ^= 1;
  }

  #pragma unroll
  for (int mt = 0; mt < 2; ++mt)
  #pragma unroll
  for (int nt = 0; nt < 2; ++nt){
    int col = bn + wn*32 + nt*16 + n;
    float bv = bias ? bias[col] : 0.f;
    #pragma unroll
    for (int r = 0; r < 8; ++r){
      int row = bm + wm*32 + mt*16 + r + 8*half;
      float v = acc[mt][nt][r] + bv;
      size_t o = (size_t)row*N + col;
      if (out_pre) out_pre[o] = f2bf(v);
      out[o] = f2bf(relu ? fmaxf(v, 0.f) : v);
    }
  }
}

// ---------------------------------------------------------------- row LayerNorm
// In-place LN over 2048 bf16 cols: v = (v-m)*rstd*sc + bi + bl  (bl = b_lstm fused)
__global__ __launch_bounds__(256) void ln_rows(
    u16* __restrict__ io, const float* __restrict__ sc,
    const float* __restrict__ bi, const float* __restrict__ bl)
{
  __shared__ float as_[8], aq_[8], mv[2];
  size_t base = (size_t)blockIdx.x * G4;
  float v[8], s = 0.f, qq = 0.f;
  #pragma unroll
  for (int i = 0; i < 8; ++i){
    int c = threadIdx.x + i*256;
    float x = bf2f(io[base + c]);
    v[i] = x; s += x; qq += x*x;
  }
  for (int m2 = 1; m2 < 32; m2 <<= 1){ s += __shfl_xor(s, m2); qq += __shfl_xor(qq, m2); }
  int wave = threadIdx.x >> 5, lane = threadIdx.x & 31;
  if (lane == 0){ as_[wave] = s; aq_[wave] = qq; }
  __syncthreads();
  if (threadIdx.x == 0){
    float S = 0.f, Q = 0.f;
    for (int i = 0; i < 8; ++i){ S += as_[i]; Q += aq_[i]; }
    float m = S * (1.f/2048.f);
    mv[0] = m; mv[1] = rsqrtf(Q*(1.f/2048.f) - m*m + LEPS);
  }
  __syncthreads();
  float m = mv[0], r = mv[1];
  #pragma unroll
  for (int i = 0; i < 8; ++i){
    int c = threadIdx.x + i*256;
    io[base + c] = f2bf((v[i] - m)*r*sc[c] + bi[c] + bl[c]);
  }
}

// ---------------------------------------------------------------- grid barrier
// Release/acquire at agent scope so cross-WGP hbuf stores are made visible
// (writeback before signal, WGP$ invalidate after the spin observes the target).
__device__ __forceinline__ void gbar(unsigned* cnt, unsigned target){
  __syncthreads();
  if (threadIdx.x == 0){
    __hip_atomic_fetch_add(cnt, 1u, __ATOMIC_RELEASE, __HIP_MEMORY_SCOPE_AGENT);
    while (__hip_atomic_load(cnt, __ATOMIC_ACQUIRE, __HIP_MEMORY_SCOPE_AGENT) < target)
      __builtin_amdgcn_s_sleep(1);
  }
  __syncthreads();
}

// ---------------------------------------------------------------- persistent LSTM
// 16 WGs x 256. WG j owns h-cols [j*32, j*32+32); wave w tile tj=j*8+w packs the
// i/f/g/o gate columns of 4 h-cols as one WMMA N-tile (custom WhF swizzle).
__global__ __launch_bounds__(256) void lstm_kernel(
    const u16* __restrict__ WhF, const u16* __restrict__ gx,
    const float* __restrict__ ghs, const float* __restrict__ ghb,
    const float* __restrict__ gcs, const float* __restrict__ gcb,
    u16* __restrict__ hbuf, u16* __restrict__ ylstm,
    float2* __restrict__ red1, float2* __restrict__ red2,
    unsigned* __restrict__ barcnt)
{
  __shared__ u16 Hs[32*520];
  __shared__ float rsum[32], rsq[32], smean[32], srstd[32];
  const int j = blockIdx.x, tid = threadIdx.x, wave = tid >> 5, lane = tid & 31;
  const int n = lane & 15, half = lane >> 4, gg = n >> 2, q = n & 3;
  const int tj = j*8 + wave;
  const int col  = gg*512 + tj*4 + q;      // gate column (0..2047)
  const int hcol = tj*4 + q;               // h column (0..511)
  const float ghs_ = ghs[col], ghb_ = ghb[col];
  const float gcs_ = gcs[hcol], gcb_ = gcb[hcol];
  float cst[16];
  #pragma unroll
  for (int v = 0; v < 16; ++v) cst[v] = 0.f;
  unsigned target = 0;

  { // zero initial h state (c starts 0 in registers)
    int gid = j*256 + tid;                 // 4096 threads x 4 u16 = 16384
    *(uint2*)&hbuf[gid*4] = make_uint2(0u, 0u);
  }
  target += 16; gbar(barcnt, target);

  for (int t = 0; t < TT; ++t){
    // stage h_t (32x512 bf16) into LDS (async DMA if available), padded pitch 520
    #pragma unroll
    for (int it = 0; it < 8; ++it){
      int c = tid + it*256;                // 2048 chunks of 16B
      int row = c >> 6, seg = c & 63;
      cp16B(&hbuf[row*512 + seg*8], &Hs[row*520 + seg*8]);
    }
    wait_async<0>();
    __syncthreads();

    // G = h @ Wh[:, wave's 16 gate cols] : M=32 (2 tiles) x N=16, K=512
    f32x8 acc0 = zero8(), acc1 = zero8();
    #pragma unroll 4
    for (int kt = 0; kt < 16; ++kt){
      size_t bo = (((size_t)tj*16 + kt)*32 + lane)*16;
      const uint4* bp = (const uint4*)(WhF + bo);
      bf16x16 bfr = mkfrag(bp[0], bp[1]);
      int base0 = n*520 + kt*32 + half*8;
      bf16x16 a0 = mkfrag(*(const uint4*)&Hs[base0], *(const uint4*)&Hs[base0 + 16]);
      int base1 = (16 + n)*520 + kt*32 + half*8;
      bf16x16 a1 = mkfrag(*(const uint4*)&Hs[base1], *(const uint4*)&Hs[base1 + 16]);
      acc0 = __builtin_amdgcn_wmma_f32_16x16x32_bf16(false, a0, false, bfr, (short)0, acc0, false, false);
      acc1 = __builtin_amdgcn_wmma_f32_16x16x32_bf16(false, a1, false, bfr, (short)0, acc1, false, false);
    }
    float G[16];
    #pragma unroll
    for (int r = 0; r < 8; ++r){ G[r] = acc0[r]; G[8 + r] = acc1[r]; }

    // gate-LN partial sums (per batch row, over this WG's 128 gate cols)
    if (tid < 32){ rsum[tid] = 0.f; rsq[tid] = 0.f; }
    __syncthreads();
    #pragma unroll
    for (int v = 0; v < 16; ++v){
      float s = G[v], qq = G[v]*G[v];
      for (int m2 = 1; m2 < 16; m2 <<= 1){ s += __shfl_xor(s, m2); qq += __shfl_xor(qq, m2); }
      if (n == 0){
        int row = ((v & 8) ? 16 : 0) + (v & 7) + 8*half;
        atomicAdd(&rsum[row], s); atomicAdd(&rsq[row], qq);
      }
    }
    __syncthreads();
    if (tid < 32) red1[j*32 + tid] = make_float2(rsum[tid], rsq[tid]);
    target += 16; gbar(barcnt, target);
    if (tid < 32){
      float s = 0.f, qq = 0.f;
      for (int jj = 0; jj < 16; ++jj){ float2 p = red1[jj*32 + tid]; s += p.x; qq += p.y; }
      float m = s * (1.f/2048.f);
      smean[tid] = m; srstd[tid] = rsqrtf(qq*(1.f/2048.f) - m*m + LEPS);
    }
    __syncthreads();

    // gates = LN(G)*ghs+ghb + gxln (gxln already holds LN(u@Wx)*gxs+gxb + b_lstm)
    float av[16];
    #pragma unroll
    for (int v = 0; v < 16; ++v){
      int row = ((v & 8) ? 16 : 0) + (v & 7) + 8*half;
      float gh = (G[v] - smean[row])*srstd[row]*ghs_ + ghb_;
      float gate = gh + bf2f(gx[((size_t)row*TT + t)*G4 + col]);
      av[v] = (gg == 2) ? tanhf(gate) : sigmoidf_(gate);
    }

    // combine i/f/g/o across lanes, update c, accumulate c-LN partials
    if (tid < 32){ rsum[tid] = 0.f; rsq[tid] = 0.f; }
    __syncthreads();
    float ov[16], cn[16];
    #pragma unroll
    for (int v = 0; v < 16; ++v){
      float i_ = av[v];
      float f_ = __shfl(av[v], lane + 4);
      float g_ = __shfl(av[v], lane + 8);
      float o_ = __shfl(av[v], lane + 12);
      float cc = f_*cst[v] + i_*g_;
      cn[v] = cc; ov[v] = o_;
      float s = (gg == 0) ? cc : 0.f, qq = (gg == 0) ? cc*cc : 0.f;
      s += __shfl_xor(s, 1); s += __shfl_xor(s, 2);
      qq += __shfl_xor(qq, 1); qq += __shfl_xor(qq, 2);
      if (n == 0){
        int row = ((v & 8) ? 16 : 0) + (v & 7) + 8*half;
        atomicAdd(&rsum[row], s); atomicAdd(&rsq[row], qq);
      }
    }
    __syncthreads();
    if (tid < 32) red2[j*32 + tid] = make_float2(rsum[tid], rsq[tid]);
    target += 16; gbar(barcnt, target);
    if (tid < 32){
      float s = 0.f, qq = 0.f;
      for (int jj = 0; jj < 16; ++jj){ float2 p = red2[jj*32 + tid]; s += p.x; qq += p.y; }
      float m = s * (1.f/512.f);
      smean[tid] = m; srstd[tid] = rsqrtf(qq*(1.f/512.f) - m*m + LEPS);
    }
    __syncthreads();

    // h = sigmoid(o) * tanh(LN(c)*gcs+gcb) ; emit h and persist state
    #pragma unroll
    for (int v = 0; v < 16; ++v){
      cst[v] = cn[v];
      if (gg == 0){
        int row = ((v & 8) ? 16 : 0) + (v & 7) + 8*half;
        float lnc = (cn[v] - smean[row])*srstd[row]*gcs_ + gcb_;
        u16 hb = f2bf(ov[v]*tanhf(lnc));
        hbuf[row*512 + hcol] = hb;
        ylstm[((size_t)row*TT + t)*512 + hcol] = hb;
      }
    }
    target += 16; gbar(barcnt, target);   // h visible before next step's stage
  }
}

// ---------------------------------------------------------------- final 512->2
__global__ void final_gemm(const u16* __restrict__ y, const float* __restrict__ Wl,
                           const float* __restrict__ bl, float* __restrict__ out){
  int idx = blockIdx.x*blockDim.x + threadIdx.x;
  if (idx >= MMR*2) return;
  int m = idx >> 1, yy = idx & 1;
  const u16* a = y + (size_t)m*512;
  float acc = bl[yy];
  for (int k = 0; k < 512; ++k) acc += bf2f(a[k]) * Wl[k*2 + yy];
  out[idx] = acc;
}

// ================================================================ host
extern "C" void kernel_launch(void* const* d_in, const int* in_sizes, int n_in,
                              void* d_out, int out_size, void* d_ws, size_t ws_size,
                              hipStream_t stream){
  (void)in_sizes; (void)n_in; (void)out_size; (void)ws_size;
  const float* x      = (const float*)d_in[0];
  const float* W_first= (const float*)d_in[1];
  const float* b_first= (const float*)d_in[2];
  const float* W1     = (const float*)d_in[3];
  const float* b1     = (const float*)d_in[4];
  const float* Wx     = (const float*)d_in[5];
  const float* Wh     = (const float*)d_in[6];
  const float* b_lstm = (const float*)d_in[7];
  const float* gx_s   = (const float*)d_in[8];
  const float* gx_b   = (const float*)d_in[9];
  const float* gh_s   = (const float*)d_in[10];
  const float* gh_b   = (const float*)d_in[11];
  const float* gc_s   = (const float*)d_in[12];
  const float* gc_b   = (const float*)d_in[13];
  const float* W2     = (const float*)d_in[14];
  const float* b2     = (const float*)d_in[15];
  const float* W3     = (const float*)d_in[16];
  const float* b3     = (const float*)d_in[17];
  const float* W_last = (const float*)d_in[18];
  const float* b_last = (const float*)d_in[19];

  size_t off = 0;
  auto alloc = [&](size_t bytes)->char*{
    off = (off + 255) & ~(size_t)255;
    char* p = (char*)d_ws + off; off += bytes; return p;
  };
  u16* WfF  = (u16*)alloc((size_t)64*512*2);
  u16* W1F  = (u16*)alloc((size_t)NBLK*1088*512*2);
  u16* WxF  = (u16*)alloc((size_t)NBLK*512*2048*2);
  u16* WhF  = (u16*)alloc((size_t)NBLK*512*2048*2);
  u16* W2F  = (u16*)alloc((size_t)NBLK*1024*512*2);
  u16* W3F  = (u16*)alloc((size_t)NBLK*512*512*2);
  u16* xbf  = (u16*)alloc((size_t)MMR*64*2);
  u16* ybf  = (u16*)alloc((size_t)MMR*512*2);
  u16* y2bf = (u16*)alloc((size_t)MMR*512*2);
  u16* ylastbf = (u16*)alloc((size_t)MMR*512*2);
  u16* ubf  = (u16*)alloc((size_t)MMR*512*2);
  u16* ylstm= (u16*)alloc((size_t)MMR*512*2);
  u16* catb = (u16*)alloc((size_t)MMR*1088*2);
  u16* gxb  = (u16*)alloc((size_t)MMR*2048*2);
  u16* hbuf = (u16*)alloc((size_t)32*512*2);
  float2* red1 = (float2*)alloc(16*32*sizeof(float2));
  float2* red2 = (float2*)alloc(16*32*sizeof(float2));
  unsigned* barcnt = (unsigned*)alloc(256);

  const int PG = 2048;
  // weight fragment packs (bf16, fragment-swizzled)
  pack_bfrag<<<PG,256,0,stream>>>(W_first, WfF, 64, 512, 512);
  for (int k = 0; k < NBLK; ++k){
    pack_bfrag<<<PG,256,0,stream>>>(W1 + (size_t)k*1088*512, W1F + (size_t)k*1088*512, 1088, 512, 512);
    pack_bfrag<<<PG,256,0,stream>>>(Wx + (size_t)k*512*2048, WxF + (size_t)k*512*2048, 512, 2048, 2048);
    pack_bfrag<<<PG,256,0,stream>>>(W2 + (size_t)k*1024*512, W2F + (size_t)k*1024*512, 1024, 512, 512);
    pack_bfrag<<<PG,256,0,stream>>>(W3 + (size_t)k*512*512,  W3F + (size_t)k*512*512,  512, 512, 512);
    pack_whfrag<<<PG,256,0,stream>>>(Wh + (size_t)k*512*2048, WhF + (size_t)k*512*2048);
  }
  cvt_bf16<<<PG,256,0,stream>>>(x, xbf, (size_t)MMR*64);

  // y = x @ W_first + b_first ; y_last = y
  gemm_wmma<<<dim3(8,128),256,0,stream>>>(xbf, WfF, b_first, ybf, nullptr, 512, 64, 0);
  (void)hipMemcpyAsync(ylastbf, ybf, (size_t)MMR*512*2, hipMemcpyDeviceToDevice, stream);

  u16* curY = ybf; u16* altY = y2bf;
  for (int k = 0; k < NBLK; ++k){
    // z = [x|y|y_last] @ W1 + b1 ; y_last = z ; u = relu(z)
    pack_cat3<<<4096,256,0,stream>>>(xbf, curY, ylastbf, catb);
    gemm_wmma<<<dim3(8,128),256,0,stream>>>(catb, W1F + (size_t)k*1088*512, b1 + k*512,
                                            ubf, ylastbf, 512, 1088, 1);
    // gx = LN(u @ Wx)*gxs+gxb + b_lstm   (precomputed x-side of gate sum)
    gemm_wmma<<<dim3(32,128),256,0,stream>>>(ubf, WxF + (size_t)k*512*2048, nullptr,
                                             gxb, nullptr, 2048, 512, 0);
    ln_rows<<<MMR,256,0,stream>>>(gxb, gx_s + k*2048, gx_b + k*2048, b_lstm + k*2048);
    // serial LSTM recurrence (persistent kernel, grid barriers)
    (void)hipMemsetAsync(barcnt, 0, sizeof(unsigned), stream);
    lstm_kernel<<<16,256,0,stream>>>(WhF + (size_t)k*512*2048, gxb,
                                     gh_s + k*2048, gh_b + k*2048,
                                     gc_s + k*512,  gc_b + k*512,
                                     hbuf, ylstm, red1, red2, barcnt);
    // y = relu([h|y_last] @ W2 + b2) ; y = relu(y @ W3 + b3)
    pack_cat2<<<4096,256,0,stream>>>(ylstm, ylastbf, catb);
    gemm_wmma<<<dim3(8,128),256,0,stream>>>(catb, W2F + (size_t)k*1024*512, b2 + k*512,
                                            curY, nullptr, 512, 1024, 1);
    gemm_wmma<<<dim3(8,128),256,0,stream>>>(curY, W3F + (size_t)k*512*512, b3 + k*512,
                                            altY, nullptr, 512, 512, 1);
    u16* t2 = curY; curY = altY; altY = t2;
  }
  final_gemm<<<(MMR*2 + 255)/256,256,0,stream>>>(curY, W_last, b_last, (float*)d_out);
}
